// QGrouping_24790551232808
// MI455X (gfx1250) — matrix operand order repaired and checked by reading, not compiled, not run on hardware
//
#include <hip/hip_runtime.h>
#include <hip/hip_bf16.h>

typedef __bf16 bf16;
typedef __bf16 v16bf __attribute__((ext_vector_type(16)));
typedef float  v8f   __attribute__((ext_vector_type(8)));
typedef float  v4f   __attribute__((ext_vector_type(4)));

#define NROWS   100000
#define IN_DIM  512
#define KEY_DIM 100
#define KGRP    8
#define VAL_DIM 64
#define NCOLS   72      // 64 val cols + 8 que cols
#define NPAD    80      // padded to 5 WMMA n-tiles of 16
#define LDSW    84      // LDS row stride (floats), breaks bank conflicts

// ---------------------------------------------------------------------------
// Kernel 0: fuse the two chained linears.
//   W_all[r][c] (bf16):  r<64 -> Wv[r][c];  64<=r<72 -> (Wq@Wk)[r-64][c];  else 0
//   b_all[r]    (f32) :  r<64 -> bv[r];     64<=r<72 -> (Wq@bk+bq)[r-64];  else 0
// ---------------------------------------------------------------------------
__global__ void build_weights(const float* __restrict__ Wk,
                              const float* __restrict__ bk,
                              const float* __restrict__ Wq,
                              const float* __restrict__ bq,
                              const float* __restrict__ Wv,
                              const float* __restrict__ bv,
                              bf16* __restrict__ Wall,
                              float* __restrict__ ball) {
    int idx = blockIdx.x * blockDim.x + threadIdx.x;
    if (idx < NPAD * IN_DIM) {
        int r = idx >> 9;       // / 512
        int c = idx & 511;
        float v = 0.f;
        if (r < VAL_DIM) {
            v = Wv[r * IN_DIM + c];
        } else if (r < NCOLS) {
            int kk = r - VAL_DIM;
            float acc = 0.f;
            for (int j = 0; j < KEY_DIM; ++j)
                acc += Wq[kk * KEY_DIM + j] * Wk[j * IN_DIM + c];
            v = acc;
        }
        Wall[idx] = (bf16)v;
    }
    if (idx < NPAD) {
        float b = 0.f;
        if (idx < VAL_DIM) {
            b = bv[idx];
        } else if (idx < NCOLS) {
            int kk = idx - VAL_DIM;
            float acc = bq[kk];
            for (int j = 0; j < KEY_DIM; ++j)
                acc += Wq[kk * KEY_DIM + j] * bk[j];
            b = acc;
        }
        ball[idx] = b;
    }
}

// ---------------------------------------------------------------------------
// Kernel 1: y[N,80] = x[N,512] @ W_all.T (bf16 WMMA, f32 accum) + b_all,
// then per-row softmax over cols 64..71 and outer product with cols 0..63.
// 8 waves/block, each wave owns one 16-row M-tile. N % 16 == 0.
// ---------------------------------------------------------------------------
__global__ __launch_bounds__(256) void qgroup_main(const float* __restrict__ x,
                                                   const bf16*  __restrict__ Wall,
                                                   const float* __restrict__ ball,
                                                   float* __restrict__ out) {
    __shared__ float lds_y[8][16][LDSW];   // 43008 B

    const int lane = threadIdx.x & 31;
    const int wave = threadIdx.x >> 5;
    const int m0   = (blockIdx.x * 8 + wave) * 16;
    if (m0 >= NROWS) return;               // wave-uniform: EXEC stays all-1s

    const int mrow = lane & 15;            // row within tile (A) / column (B,C)
    const int hi   = lane >> 4;            // lane-half select per ISA layouts

    v8f acc[5];
    #pragma unroll
    for (int n = 0; n < 5; ++n)
        #pragma unroll
        for (int r = 0; r < 8; ++r) acc[n][r] = 0.f;

    const float* xrow = x + (size_t)(m0 + mrow) * IN_DIM;
    const bf16*  wrow = Wall + (size_t)mrow * IN_DIM;

    for (int k0 = 0; k0 < IN_DIM; k0 += 32) {
        // ---- A fragment: 16x32 bf16. Layout: lanes 0-15 hold K 0..15,
        // lanes 16-31 hold K 16..31 for row M = lane&15 (matches B).
        const int ka = k0 + hi * 16;
        v4f a0 = *reinterpret_cast<const v4f*>(xrow + ka);
        v4f a1 = *reinterpret_cast<const v4f*>(xrow + ka + 4);
        v4f a2 = *reinterpret_cast<const v4f*>(xrow + ka + 8);
        v4f a3 = *reinterpret_cast<const v4f*>(xrow + ka + 12);
        v16bf A;
        #pragma unroll
        for (int i = 0; i < 4; ++i) {
            A[i]      = (bf16)a0[i];
            A[4 + i]  = (bf16)a1[i];
            A[8 + i]  = (bf16)a2[i];
            A[12 + i] = (bf16)a3[i];
        }
        // ---- B fragments: 32x16 bf16. Layout used here (contiguous k, same as A): lane = col (lane&15),
        // lanes 0-15 hold K 0..15, lanes 16-31 hold K 16..31 -> one contiguous
        // 32-byte read from row-major W_all (col of W_all.T == row of W_all).
        // Batch all 5 loads first so the clause overlaps latency, then issue
        // the 5 WMMAs with staggered waits instead of load->wait0->wmma chains.
        const int kb = k0 + hi * 16;
        v16bf B[5];
        #pragma unroll
        for (int n = 0; n < 5; ++n)
            B[n] = *reinterpret_cast<const v16bf*>(wrow + (size_t)(n * 16) * IN_DIM + kb);
        #pragma unroll
        for (int n = 0; n < 5; ++n)
            acc[n] = __builtin_amdgcn_wmma_f32_16x16x32_bf16(
                false, A, false, B[n], (short)0, acc[n], false, false);
    }

    // ---- spill y tile (+ bias) to this wave's private LDS slab.
    // C layout: VGPR r holds (M = r + 8*hi, Ncol = lane&15).
    #pragma unroll
    for (int n = 0; n < 5; ++n) {
        float b = ball[n * 16 + mrow];
        #pragma unroll
        for (int r = 0; r < 8; ++r)
            lds_y[wave][r + hi * 8][n * 16 + mrow] = acc[n][r] + b;
    }
    asm volatile("s_wait_dscnt 0" ::: "memory");   // wave-private RAW fence

    // ---- softmax over que (cols 64..71) + outer product with val (cols 0..63)
    for (int m = 0; m < 16; ++m) {
        float qv[8];
        #pragma unroll
        for (int kk = 0; kk < 8; ++kk) qv[kk] = lds_y[wave][m][64 + kk];
        float mx = qv[0];
        #pragma unroll
        for (int kk = 1; kk < 8; ++kk) mx = fmaxf(mx, qv[kk]);
        float w[8], s = 0.f;
        #pragma unroll
        for (int kk = 0; kk < 8; ++kk) { w[kk] = __expf(qv[kk] - mx); s += w[kk]; }
        const float inv = 1.f / s;
        #pragma unroll
        for (int kk = 0; kk < 8; ++kk) w[kk] *= inv;

        // each lane reuses one 4-float slice of val
        v4f val = *reinterpret_cast<const v4f*>(&lds_y[wave][m][(lane & 15) * 4]);
        float* orow = out + (size_t)(m0 + m) * (KGRP * VAL_DIM);
        #pragma unroll
        for (int i = 0; i < 4; ++i) {
            // flat index = (lane + 32*i)*4 ; kk = 2*i + hi (constant-indexed)
            const float wk = hi ? w[2 * i + 1] : w[2 * i];
            v4f o = { wk * val[0], wk * val[1], wk * val[2], wk * val[3] };
            *reinterpret_cast<v4f*>(orow + (lane + 32 * i) * 4) = o;  // 512B/wave, coalesced
        }
    }
}

// ---------------------------------------------------------------------------
extern "C" void kernel_launch(void* const* d_in, const int* in_sizes, int n_in,
                              void* d_out, int out_size, void* d_ws, size_t ws_size,
                              hipStream_t stream) {
    (void)in_sizes; (void)n_in; (void)out_size; (void)ws_size;
    const float* x  = (const float*)d_in[0];
    const float* Wk = (const float*)d_in[1];
    const float* bk = (const float*)d_in[2];
    const float* Wq = (const float*)d_in[3];
    const float* bq = (const float*)d_in[4];
    const float* Wv = (const float*)d_in[5];
    const float* bv = (const float*)d_in[6];
    float* out = (float*)d_out;

    bf16*  Wall = (bf16*)d_ws;
    float* ball = (float*)((char*)d_ws + (size_t)NPAD * IN_DIM * sizeof(bf16));

    build_weights<<<(NPAD * IN_DIM + 255) / 256, 256, 0, stream>>>(
        Wk, bk, Wq, bq, Wv, bv, Wall, ball);

    const int nblocks = (NROWS + 127) / 128;   // 782; trailing waves self-disable
    qgroup_main<<<nblocks, 256, 0, stream>>>(x, Wall, ball, out);
}
